// MyViTBlock_82016695484523
// MI455X (gfx1250) — compile-verified, matrix operand order checked by reading, not run on hardware
//
#include <hip/hip_runtime.h>

typedef __attribute__((ext_vector_type(16))) _Float16 v16h;
typedef __attribute__((ext_vector_type(8)))  _Float16 v8h;
typedef __attribute__((ext_vector_type(8)))  float    v8f;

#define BATCH   32
#define S_LEN   577
#define S_PAD   608      // 19 * 32, K-padding for P@V
#define NT_S    37       // ceil(577/16)
#define D_MODEL 768
#define NHEAD   12
#define DHD     64
#define MLP_DIM 3072
#define M_ROWS  (BATCH * S_LEN)   // 18464 = 577 * 32
#define MT32    577

// ---------------------------------------------------------------------------
// Fragment helpers. CDNA5 16-bit A-matrix 16x32 layout (ISA 7.12.2):
//   lane L: row m = L%16, half = L/16
//   element e<8 : k = kbase + 8*half + e
//   element e>=8: k = kbase + 16 + 8*half + (e-8)
// For row-major K-contiguous storage this is two 16-byte loads per lane.
// B matrix (K x 16) uses the identical pattern when stored as [n][k] with k
// contiguous — exactly the torch Linear weight layout [out,in].
// ---------------------------------------------------------------------------
__device__ __forceinline__ v16h load_frag(const _Float16* rowbase, int kofs) {
  v8h lo = *(const v8h*)(rowbase + kofs);
  v8h hi = *(const v8h*)(rowbase + kofs + 16);
  v16h f;
#pragma unroll
  for (int i = 0; i < 8; ++i) { f[i] = lo[i]; f[i + 8] = hi[i]; }
  return f;
}

__device__ __forceinline__ v16h zero_frag() {
  v16h f;
#pragma unroll
  for (int i = 0; i < 16; ++i) f[i] = (_Float16)0.f;
  return f;
}

__device__ __forceinline__ v8f wmma_f16(v16h a, v16h b, v8f c) {
  return __builtin_amdgcn_wmma_f32_16x16x32_f16(false, a, false, b, (short)0, c,
                                                false, false);
}

// ---------------------------------------------------------------------------
// f32 -> f16 weight conversion
// ---------------------------------------------------------------------------
__global__ void cvt_f2h(const float* __restrict__ in, _Float16* __restrict__ out, int n) {
  int i = blockIdx.x * blockDim.x + threadIdx.x;
  if (i < n) out[i] = (_Float16)in[i];
}

// ---------------------------------------------------------------------------
// LayerNorm over D=768, one block (256 thr) per row; f32 in, f16 out
// ---------------------------------------------------------------------------
__global__ __launch_bounds__(256) void ln_kernel(const float* __restrict__ x,
                                                 const float* __restrict__ g,
                                                 const float* __restrict__ bt,
                                                 _Float16* __restrict__ out) {
  int row = blockIdx.x;
  const float* xr = x + (size_t)row * D_MODEL;
  int tid = threadIdx.x;
  float v[3], s = 0.f, ss = 0.f;
#pragma unroll
  for (int j = 0; j < 3; ++j) {
    float t = xr[tid + j * 256];
    v[j] = t; s += t; ss += t * t;
  }
#pragma unroll
  for (int o = 16; o > 0; o >>= 1) {
    s  += __shfl_xor(s, o, 32);
    ss += __shfl_xor(ss, o, 32);
  }
  __shared__ float sh[2][8];
  int wid = tid >> 5, lane = tid & 31;
  if (lane == 0) { sh[0][wid] = s; sh[1][wid] = ss; }
  __syncthreads();
  if (tid == 0) {
    float a = 0.f, b2 = 0.f;
    for (int i = 0; i < 8; ++i) { a += sh[0][i]; b2 += sh[1][i]; }
    float mu = a / D_MODEL;
    float var = b2 / D_MODEL - mu * mu;
    sh[0][0] = mu;
    sh[1][0] = rsqrtf(var + 1e-5f);
  }
  __syncthreads();
  float mu = sh[0][0], rinv = sh[1][0];
  _Float16* outr = out + (size_t)row * D_MODEL;
#pragma unroll
  for (int j = 0; j < 3; ++j) {
    int c = tid + j * 256;
    outr[c] = (_Float16)((v[j] - mu) * rinv * g[c] + bt[c]);
  }
}

// ---------------------------------------------------------------------------
// Per-head QKV projection: [S,64] @ [64,64]^T (+bias).
// grid (37, 12, 32), block 128 = 4 waves; wave w owns n-tile w.
// q,k stored [B,H,S,64]; v stored transposed [B,H,64,S_PAD] zero-padded.
// ---------------------------------------------------------------------------
__global__ __launch_bounds__(128) void qkv_kernel(
    const _Float16* __restrict__ xn,
    const _Float16* __restrict__ Wq, const float* __restrict__ bq,
    const _Float16* __restrict__ Wk, const float* __restrict__ bk,
    const _Float16* __restrict__ Wv, const float* __restrict__ bv,
    _Float16* __restrict__ q, _Float16* __restrict__ k, _Float16* __restrict__ vT) {
  int st = blockIdx.x, h = blockIdx.y, b = blockIdx.z;
  int w = threadIdx.x >> 5, lane = threadIdx.x & 31;
  int kh = lane >> 4, lidx = lane & 15;
  int n = w * 16 + lidx;

  const _Float16* wqb = Wq + ((size_t)h * DHD + n) * DHD;
  const _Float16* wkb = Wk + ((size_t)h * DHD + n) * DHD;
  const _Float16* wvb = Wv + ((size_t)h * DHD + n) * DHD;

  int m = st * 16 + lidx;
  bool avalid = (m < S_LEN);
  const _Float16* arow = xn + ((size_t)b * S_LEN + (avalid ? m : 0)) * D_MODEL + h * DHD;

  v8f accq = {}, acck = {}, accv = {};
#pragma unroll
  for (int ks = 0; ks < 2; ++ks) {
    int kofs = ks * 32 + 8 * kh;
    v16h a  = avalid ? load_frag(arow, kofs) : zero_frag();
    v16h fq = load_frag(wqb, kofs);
    v16h fk = load_frag(wkb, kofs);
    v16h fv = load_frag(wvb, kofs);
    accq = wmma_f16(a, fq, accq);
    acck = wmma_f16(a, fk, acck);
    accv = wmma_f16(a, fv, accv);
  }
  float bqv = bq[h * DHD + n], bkv = bk[h * DHD + n], bvv = bv[h * DHD + n];
  size_t bh = (size_t)b * NHEAD + h;
#pragma unroll
  for (int r = 0; r < 8; ++r) {
    int s = st * 16 + r + 8 * kh;
    if (s < S_LEN) {
      q[(bh * S_LEN + s) * DHD + n]  = (_Float16)(accq[r] + bqv);
      k[(bh * S_LEN + s) * DHD + n]  = (_Float16)(acck[r] + bkv);
      vT[(bh * DHD + n) * S_PAD + s] = (_Float16)(accv[r] + bvv);
    } else {
      vT[(bh * DHD + n) * S_PAD + s] = (_Float16)0.f;
    }
  }
  if (st == NT_S - 1) {
#pragma unroll
    for (int r = 0; r < 8; ++r) {
      int s = 592 + r + 8 * kh;
      vT[(bh * DHD + n) * S_PAD + s] = (_Float16)0.f;
    }
  }
}

// ---------------------------------------------------------------------------
// Attention for one (b, h, 16-query tile). 256 thr = 8 waves.
// Phase 1: scores (Q.K^T / 8) -> f32 LDS [16][608]
// Phase 2: row softmax -> f16 probs in LDS (A-matrix layout source)
// Phase 3: P @ V^T via WMMA (A from LDS, B from global vT), out = x + attn
// ---------------------------------------------------------------------------
__global__ __launch_bounds__(256) void attn_kernel(
    const _Float16* __restrict__ q, const _Float16* __restrict__ k,
    const _Float16* __restrict__ vT, const float* __restrict__ x,
    float* __restrict__ out) {
  __shared__ __align__(16) float    Sc[16][S_PAD];
  __shared__ __align__(16) _Float16 Pp[16][S_PAD];
  __shared__ float red16[16][16];
  __shared__ float rstat[2][16];

  int qt = blockIdx.x, h = blockIdx.y, b = blockIdx.z;
  size_t bh = (size_t)b * NHEAD + h;
  int tid = threadIdx.x;
  int w = tid >> 5, lane = tid & 31, kh = lane >> 4, lidx = lane & 15;

  const _Float16* qb = q + bh * S_LEN * DHD;
  const _Float16* kb = k + bh * S_LEN * DHD;

  int mrow = qt * 16 + lidx;
  bool mval = mrow < S_LEN;
  const _Float16* arow = qb + (size_t)(mval ? mrow : 0) * DHD;
  v16h qa0 = mval ? load_frag(arow, 8 * kh)      : zero_frag();
  v16h qa1 = mval ? load_frag(arow, 32 + 8 * kh) : zero_frag();

  for (int kt = w; kt < NT_S; kt += 8) {
    int nrow = kt * 16 + lidx;
    bool nval = nrow < S_LEN;
    const _Float16* brow = kb + (size_t)(nval ? nrow : 0) * DHD;
    v16h b0 = nval ? load_frag(brow, 8 * kh)      : zero_frag();
    v16h b1 = nval ? load_frag(brow, 32 + 8 * kh) : zero_frag();
    v8f acc = {};
    acc = wmma_f16(qa0, b0, acc);
    acc = wmma_f16(qa1, b1, acc);
#pragma unroll
    for (int r = 0; r < 8; ++r)
      Sc[r + 8 * kh][kt * 16 + lidx] = acc[r] * 0.125f;   // 1/sqrt(64)
  }
  __syncthreads();

  // softmax: 16 threads per row
  int row = tid >> 4, t = tid & 15;
  float lmax = -3.0e38f;
  for (int c = t; c < S_LEN; c += 16) lmax = fmaxf(lmax, Sc[row][c]);
  red16[row][t] = lmax;
  __syncthreads();
  if (t == 0) {
    float mv = red16[row][0];
    for (int i = 1; i < 16; ++i) mv = fmaxf(mv, red16[row][i]);
    rstat[0][row] = mv;
  }
  __syncthreads();
  float rmax = rstat[0][row];
  float lsum = 0.f;
  for (int c = t; c < S_LEN; c += 16) lsum += __expf(Sc[row][c] - rmax);
  red16[row][t] = lsum;
  __syncthreads();
  if (t == 0) {
    float sv = 0.f;
    for (int i = 0; i < 16; ++i) sv += red16[row][i];
    rstat[1][row] = sv;
  }
  __syncthreads();
  float rinv = 1.f / rstat[1][row];
  for (int c = t; c < S_PAD; c += 16)
    Pp[row][c] = (c < S_LEN) ? (_Float16)(__expf(Sc[row][c] - rmax) * rinv)
                             : (_Float16)0.f;
  __syncthreads();

  if (w < 4) {
    int n = w * 16 + lidx;
    const _Float16* vrow = vT + (bh * DHD + n) * S_PAD;
    const _Float16* prow = &Pp[lidx][0];
    v8f acc = {};
#pragma unroll 1
    for (int ks = 0; ks < S_PAD / 32; ++ks) {
      int kofs = ks * 32 + 8 * kh;
      acc = wmma_f16(load_frag(prow, kofs), load_frag(vrow, kofs), acc);
    }
#pragma unroll
    for (int r = 0; r < 8; ++r) {
      int s = qt * 16 + r + 8 * kh;
      if (s < S_LEN) {
        size_t idx = ((size_t)b * S_LEN + s) * D_MODEL + h * DHD + n;
        out[idx] = x[idx] + acc[r];
      }
    }
  }
}

// ---------------------------------------------------------------------------
// MLP1: [18464,768] @ [3072,768]^T + b1, exact GELU, f16 out.
// grid (577, 6), 256 thr = 8 waves. Wave tile = 32M x 64N:
//   2 A-frags x 4 B-frags -> 8 WMMA per 32-K step (~21 flops/byte).
// ---------------------------------------------------------------------------
__global__ __launch_bounds__(256) void mlp1_kernel(
    const _Float16* __restrict__ hx, const _Float16* __restrict__ W1,
    const float* __restrict__ b1, _Float16* __restrict__ mo) {
  int mt = blockIdx.x, nc = blockIdx.y;
  int w = threadIdx.x >> 5, lane = threadIdx.x & 31;
  int kh = lane >> 4, lidx = lane & 15;
  int nbase = nc * 512 + w * 64;
  int mbase = mt * 32;
  const _Float16* arow0 = hx + (size_t)(mbase + lidx) * D_MODEL;
  const _Float16* arow1 = hx + (size_t)(mbase + 16 + lidx) * D_MODEL;
  const _Float16* brow0 = W1 + (size_t)(nbase + lidx) * D_MODEL;
  const _Float16* brow1 = brow0 + (size_t)16 * D_MODEL;
  const _Float16* brow2 = brow0 + (size_t)32 * D_MODEL;
  const _Float16* brow3 = brow0 + (size_t)48 * D_MODEL;

  v8f acc[2][4] = {};
#pragma unroll 1
  for (int ks = 0; ks < D_MODEL / 32; ++ks) {
    int kofs = ks * 32 + 8 * kh;
    __builtin_prefetch(arow0 + kofs + 128, 0, 1);  // global_prefetch_b8
    __builtin_prefetch(arow1 + kofs + 128, 0, 1);
    v16h a0 = load_frag(arow0, kofs);
    v16h a1 = load_frag(arow1, kofs);
    v16h b0 = load_frag(brow0, kofs);
    v16h b1f = load_frag(brow1, kofs);
    v16h b2f = load_frag(brow2, kofs);
    v16h b3f = load_frag(brow3, kofs);
    acc[0][0] = wmma_f16(a0, b0,  acc[0][0]);
    acc[0][1] = wmma_f16(a0, b1f, acc[0][1]);
    acc[0][2] = wmma_f16(a0, b2f, acc[0][2]);
    acc[0][3] = wmma_f16(a0, b3f, acc[0][3]);
    acc[1][0] = wmma_f16(a1, b0,  acc[1][0]);
    acc[1][1] = wmma_f16(a1, b1f, acc[1][1]);
    acc[1][2] = wmma_f16(a1, b2f, acc[1][2]);
    acc[1][3] = wmma_f16(a1, b3f, acc[1][3]);
  }
#pragma unroll
  for (int j = 0; j < 4; ++j) {
    int n = nbase + j * 16 + lidx;
    float bias = b1[n];
#pragma unroll
    for (int mi = 0; mi < 2; ++mi) {
#pragma unroll
      for (int r = 0; r < 8; ++r) {
        int mr = mbase + mi * 16 + r + 8 * kh;
        float v = acc[mi][j][r] + bias;
        float gl = 0.5f * v * (1.f + erff(v * 0.70710678118f));
        mo[(size_t)mr * MLP_DIM + n] = (_Float16)gl;
      }
    }
  }
}

// ---------------------------------------------------------------------------
// MLP2: [18464,3072] @ [768,3072]^T + b2, accumulate into out (residual).
// grid (577, 3), 128 thr = 4 waves. Wave tile = 32M x 64N.
// ---------------------------------------------------------------------------
__global__ __launch_bounds__(128) void mlp2_kernel(
    const _Float16* __restrict__ mi_, const _Float16* __restrict__ W2,
    const float* __restrict__ b2, float* __restrict__ out) {
  int mt = blockIdx.x, nc = blockIdx.y;
  int w = threadIdx.x >> 5, lane = threadIdx.x & 31;
  int kh = lane >> 4, lidx = lane & 15;
  int nbase = nc * 256 + w * 64;
  int mbase = mt * 32;
  const _Float16* arow0 = mi_ + (size_t)(mbase + lidx) * MLP_DIM;
  const _Float16* arow1 = mi_ + (size_t)(mbase + 16 + lidx) * MLP_DIM;
  const _Float16* brow0 = W2 + (size_t)(nbase + lidx) * MLP_DIM;
  const _Float16* brow1 = brow0 + (size_t)16 * MLP_DIM;
  const _Float16* brow2 = brow0 + (size_t)32 * MLP_DIM;
  const _Float16* brow3 = brow0 + (size_t)48 * MLP_DIM;

  v8f acc[2][4] = {};
#pragma unroll 1
  for (int ks = 0; ks < MLP_DIM / 32; ++ks) {
    int kofs = ks * 32 + 8 * kh;
    __builtin_prefetch(arow0 + kofs + 128, 0, 1);
    __builtin_prefetch(arow1 + kofs + 128, 0, 1);
    v16h a0 = load_frag(arow0, kofs);
    v16h a1 = load_frag(arow1, kofs);
    v16h b0 = load_frag(brow0, kofs);
    v16h b1f = load_frag(brow1, kofs);
    v16h b2f = load_frag(brow2, kofs);
    v16h b3f = load_frag(brow3, kofs);
    acc[0][0] = wmma_f16(a0, b0,  acc[0][0]);
    acc[0][1] = wmma_f16(a0, b1f, acc[0][1]);
    acc[0][2] = wmma_f16(a0, b2f, acc[0][2]);
    acc[0][3] = wmma_f16(a0, b3f, acc[0][3]);
    acc[1][0] = wmma_f16(a1, b0,  acc[1][0]);
    acc[1][1] = wmma_f16(a1, b1f, acc[1][1]);
    acc[1][2] = wmma_f16(a1, b2f, acc[1][2]);
    acc[1][3] = wmma_f16(a1, b3f, acc[1][3]);
  }
#pragma unroll
  for (int j = 0; j < 4; ++j) {
    int n = nbase + j * 16 + lidx;
    float bias = b2[n];
#pragma unroll
    for (int mi = 0; mi < 2; ++mi) {
#pragma unroll
      for (int r = 0; r < 8; ++r) {
        int mr = mbase + mi * 16 + r + 8 * kh;
        size_t idx = (size_t)mr * D_MODEL + n;
        out[idx] = out[idx] + acc[mi][j][r] + bias;
      }
    }
  }
}

// ---------------------------------------------------------------------------
extern "C" void kernel_launch(void* const* d_in, const int* in_sizes, int n_in,
                              void* d_out, int out_size, void* d_ws, size_t ws_size,
                              hipStream_t stream) {
  (void)in_sizes; (void)n_in; (void)out_size; (void)ws_size;
  const float* x    = (const float*)d_in[0];
  const float* ln1g = (const float*)d_in[1];
  const float* ln1b = (const float*)d_in[2];
  const float* ln2g = (const float*)d_in[3];
  const float* ln2b = (const float*)d_in[4];
  const float* Wq   = (const float*)d_in[5];
  const float* bq   = (const float*)d_in[6];
  const float* Wk   = (const float*)d_in[7];
  const float* bk   = (const float*)d_in[8];
  const float* Wv   = (const float*)d_in[9];
  const float* bv   = (const float*)d_in[10];
  const float* W1   = (const float*)d_in[11];
  const float* b1   = (const float*)d_in[12];
  const float* W2   = (const float*)d_in[13];
  const float* b2   = (const float*)d_in[14];
  float* out = (float*)d_out;

  char* p = (char*)d_ws;
  auto alloc_h = [&](size_t nelem) {
    _Float16* r = (_Float16*)p;
    p += ((nelem * sizeof(_Float16) + 255) / 256) * 256;
    return r;
  };
  const size_t WQKV = (size_t)NHEAD * DHD * DHD;       // 49152
  const size_t WBIG = (size_t)MLP_DIM * D_MODEL;       // 2359296
  _Float16* Wq_h = alloc_h(WQKV);
  _Float16* Wk_h = alloc_h(WQKV);
  _Float16* Wv_h = alloc_h(WQKV);
  _Float16* W1_h = alloc_h(WBIG);
  _Float16* W2_h = alloc_h(WBIG);
  _Float16* xn_h = alloc_h((size_t)M_ROWS * D_MODEL);
  _Float16* q_h  = alloc_h((size_t)BATCH * NHEAD * S_LEN * DHD);
  _Float16* k_h  = alloc_h((size_t)BATCH * NHEAD * S_LEN * DHD);
  _Float16* vT_h = alloc_h((size_t)BATCH * NHEAD * DHD * S_PAD);
  _Float16* h_h  = alloc_h((size_t)M_ROWS * D_MODEL);
  _Float16* m_h  = alloc_h((size_t)M_ROWS * MLP_DIM);

  auto cvt = [&](const float* src, _Float16* dst, size_t n) {
    cvt_f2h<<<(unsigned)((n + 255) / 256), 256, 0, stream>>>(src, dst, (int)n);
  };
  cvt(Wq, Wq_h, WQKV);
  cvt(Wk, Wk_h, WQKV);
  cvt(Wv, Wv_h, WQKV);
  cvt(W1, W1_h, WBIG);
  cvt(W2, W2_h, WBIG);

  ln_kernel<<<M_ROWS, 256, 0, stream>>>(x, ln1g, ln1b, xn_h);

  qkv_kernel<<<dim3(NT_S, NHEAD, BATCH), 128, 0, stream>>>(
      xn_h, Wq_h, bq, Wk_h, bk, Wv_h, bv, q_h, k_h, vT_h);

  attn_kernel<<<dim3(NT_S, NHEAD, BATCH), 256, 0, stream>>>(q_h, k_h, vT_h, x, out);

  ln_kernel<<<M_ROWS, 256, 0, stream>>>(out, ln2g, ln2b, h_h);

  mlp1_kernel<<<dim3(MT32, MLP_DIM / 512), 256, 0, stream>>>(h_h, W1_h, b1, m_h);

  mlp2_kernel<<<dim3(MT32, D_MODEL / 256), 128, 0, stream>>>(m_h, W2_h, b2, out);
}
